// BinaryLinear_81063212745415
// MI455X (gfx1250) — compile-verified, matrix operand order checked by reading, not run on hardware
//
#include <hip/hip_runtime.h>

// ---------------------------------------------------------------------------
// BinaryLinear: out[m, o] = sum_i x[m,i] * sign(W[o,i]) + bias[o]
// M = 16384, N = 4096, K = 4096.
// Precision: x = hi + lo Dekker split in f16 (W = +-1 exact in f16);
//            acc += hi*W; acc += lo*W with V_WMMA_F32_16X16X32_F16 -> ~fp32.
// Modes (chosen by ws_size, deterministic):
//   FULL  : x pre-split to f16 hi/lo + W prebinarized in d_ws; GEMM stages
//           A-hi/A-lo/B with global_load_async_to_lds_b128 (ASYNCcnt DMA),
//           near-zero VALU in the main loop.
//   PREBIN: W prebinarized only; x split fused in the staging path.
//   FUSED : no workspace; everything converted on the fly.
// 128x256 block tile, 8 waves, 64x64 per wave = 32 WMMA / K-step / wave.
// ---------------------------------------------------------------------------

typedef _Float16 v4h  __attribute__((ext_vector_type(4)));
typedef _Float16 v8h  __attribute__((ext_vector_type(8)));
typedef _Float16 v16h __attribute__((ext_vector_type(16)));
typedef float    v8f  __attribute__((ext_vector_type(8)));

#define K_DIM   4096
#define N_DIM   4096
#define TILE_M  128
#define TILE_N  256
#define TILE_K  32
#define LDSK    40   // padded K stride (f16): 80B = 20 banks -> conflict-free, 16B aligned

#define MODE_FUSED  0
#define MODE_PREBIN 1
#define MODE_FULL   2

// ---------------- W prebinarization: fp32 -> f16 {+1,-1} --------------------
__global__ __launch_bounds__(256) void binarize_w(const float* __restrict__ w,
                                                  _Float16* __restrict__ wb) {
    const int i = (blockIdx.x * 256 + threadIdx.x) * 8;
    const float4 a = *(const float4*)(w + i);
    const float4 b = *(const float4*)(w + i + 4);
    const float f[8] = {a.x, a.y, a.z, a.w, b.x, b.y, b.z, b.w};
    v8h o;
    #pragma unroll
    for (int e = 0; e < 8; ++e)
        o[e] = (f[e] >= 0.0f) ? (_Float16)1.0f : (_Float16)-1.0f;
    *(v8h*)(wb + i) = o;
}

// ---------------- x Dekker split: fp32 -> f16 hi + f16 lo -------------------
__global__ __launch_bounds__(256) void split_x(const float* __restrict__ x,
                                               _Float16* __restrict__ xhi,
                                               _Float16* __restrict__ xlo) {
    const int i = (blockIdx.x * 256 + threadIdx.x) * 8;
    const float4 a = *(const float4*)(x + i);
    const float4 b = *(const float4*)(x + i + 4);
    const float f[8] = {a.x, a.y, a.z, a.w, b.x, b.y, b.z, b.w};
    v8h hi, lo;
    #pragma unroll
    for (int e = 0; e < 8; ++e) {
        const _Float16 h = (_Float16)f[e];        // RNE
        hi[e] = h;
        lo[e] = (_Float16)(f[e] - (float)h);      // residual: ~fp32 accuracy combined
    }
    *(v8h*)(xhi + i) = hi;
    *(v8h*)(xlo + i) = lo;
}

// ------------------------------- GEMM ---------------------------------------
template <int MODE>
__global__ __launch_bounds__(256) void binlin_wmma(
    const float*    __restrict__ x,      // [M, K] fp32
    const float*    __restrict__ wfp,    // [N, K] fp32          (FUSED)
    const _Float16* __restrict__ wbin,   // [N, K] f16 +-1       (PREBIN/FULL)
    const _Float16* __restrict__ xhi,    // [M, K] f16 hi        (FULL)
    const _Float16* __restrict__ xlo,    // [M, K] f16 lo        (FULL)
    const float*    __restrict__ bias,   // [N]
    float*          __restrict__ out)    // [M, N]
{
    __shared__ alignas(16) _Float16 sAhi[2][TILE_M * LDSK];
    __shared__ alignas(16) _Float16 sAlo[2][TILE_M * LDSK];
    __shared__ alignas(16) _Float16 sB  [2][TILE_N * LDSK];

    const int tid   = threadIdx.x;
    const int lane  = tid & 31;
    const int wave  = tid >> 5;       // 0..7
    const int waveM = wave >> 2;      // 0..1 -> 64-row slab
    const int waveN = wave & 3;       // 0..3 -> 64-col slab

    const int mBase = blockIdx.x * TILE_M;   // M fastest: consecutive blocks share B in L2
    const int nBase = blockIdx.y * TILE_N;

    const float*    xTile = x + (size_t)mBase * K_DIM;
    const float*    wTile = (MODE == MODE_FUSED) ? (wfp  + (size_t)nBase * K_DIM) : nullptr;
    const _Float16* bTile = (MODE != MODE_FUSED) ? (wbin + (size_t)nBase * K_DIM) : nullptr;
    const _Float16* hTile = (MODE == MODE_FULL)  ? (xhi  + (size_t)mBase * K_DIM) : nullptr;
    const _Float16* lTile = (MODE == MODE_FULL)  ? (xlo  + (size_t)mBase * K_DIM) : nullptr;

    float4 aR[4];
    float4 bR[8];

    // ---- A staging, FULL: async DMA of pre-split hi/lo (no VALU) ----
    auto asyncA = [&](int buf, int k0) {
        #pragma unroll
        for (int i = 0; i < 2; ++i) {
            const int idx = tid + (i << 8);
            const int r   = idx >> 2;          // 0..127
            const int c   = (idx & 3) << 3;    // 0,8,16,24 (f16 elems)
            const _Float16* gh = hTile + (size_t)r * K_DIM + (k0 + c);
            const _Float16* gl = lTile + (size_t)r * K_DIM + (k0 + c);
            const unsigned oh = (unsigned)(size_t)&sAhi[buf][r * LDSK + c];
            const unsigned ol = (unsigned)(size_t)&sAlo[buf][r * LDSK + c];
            asm volatile("global_load_async_to_lds_b128 %0, %1, off"
                         :: "v"(oh), "v"(gh) : "memory");
            asm volatile("global_load_async_to_lds_b128 %0, %1, off"
                         :: "v"(ol), "v"(gl) : "memory");
        }
    };
    // ---- A staging, FUSED/PREBIN: load fp32, split in registers ----
    auto loadA = [&](int k0) {
        #pragma unroll
        for (int i = 0; i < 4; ++i) {
            const int idx = tid + (i << 8);
            const int r   = idx >> 3;          // 0..127
            const int c   = (idx & 7) << 2;    // 0..28
            aR[i] = *(const float4*)(xTile + (size_t)r * K_DIM + (k0 + c));
        }
    };
    auto stageA = [&](int buf) {
        #pragma unroll
        for (int i = 0; i < 4; ++i) {
            const int idx = tid + (i << 8);
            const int r   = idx >> 3;
            const int c   = (idx & 7) << 2;
            const float f[4] = {aR[i].x, aR[i].y, aR[i].z, aR[i].w};
            v4h hi, lo;
            #pragma unroll
            for (int e = 0; e < 4; ++e) {
                const _Float16 h = (_Float16)f[e];
                hi[e] = h;
                lo[e] = (_Float16)(f[e] - (float)h);
            }
            *(v4h*)&sAhi[buf][r * LDSK + c] = hi;
            *(v4h*)&sAlo[buf][r * LDSK + c] = lo;
        }
    };

    // ---- B staging, PREBIN/FULL: async DMA of prebinarized f16 ----
    auto asyncB = [&](int buf, int k0) {
        #pragma unroll
        for (int i = 0; i < 4; ++i) {
            const int idx = tid + (i << 8);
            const int r   = idx >> 2;          // 0..255
            const int c   = (idx & 3) << 3;
            const _Float16* g = bTile + (size_t)r * K_DIM + (k0 + c);
            const unsigned ldsOff = (unsigned)(size_t)&sB[buf][r * LDSK + c];
            asm volatile("global_load_async_to_lds_b128 %0, %1, off"
                         :: "v"(ldsOff), "v"(g) : "memory");
        }
    };
    // ---- B staging, FUSED: binarize fp32 on the fly ----
    auto loadB = [&](int k0) {
        #pragma unroll
        for (int i = 0; i < 8; ++i) {
            const int idx = tid + (i << 8);
            const int r   = idx >> 3;          // 0..255
            const int c   = (idx & 7) << 2;
            bR[i] = *(const float4*)(wTile + (size_t)r * K_DIM + (k0 + c));
        }
    };
    auto stageB = [&](int buf) {
        #pragma unroll
        for (int i = 0; i < 8; ++i) {
            const int idx = tid + (i << 8);
            const int r   = idx >> 3;
            const int c   = (idx & 7) << 2;
            const float f[4] = {bR[i].x, bR[i].y, bR[i].z, bR[i].w};
            v4h bb;
            #pragma unroll
            for (int e = 0; e < 4; ++e)
                bb[e] = (f[e] >= 0.0f) ? (_Float16)1.0f : (_Float16)-1.0f;
            *(v4h*)&sB[buf][r * LDSK + c] = bb;
        }
    };

    v8f acc[4][4] = {};

    // ---- prologue ----
    if (MODE == MODE_FULL) {
        asyncA(0, 0);
        asyncB(0, 0);
        asm volatile("s_wait_asynccnt 0x0" ::: "memory");
    } else {
        loadA(0);
        if (MODE == MODE_PREBIN) asyncB(0, 0); else loadB(0);
        stageA(0);
        if (MODE == MODE_FUSED) stageB(0);
        if (MODE == MODE_PREBIN) asm volatile("s_wait_asynccnt 0x0" ::: "memory");
    }
    __syncthreads();

    const int KSTEPS = K_DIM / TILE_K;   // 128
    const int rsel = lane & 15;          // row within 16x16 tile
    const int kb   = (lane >> 4) << 3;   // K sub-group base: 0 or 8 (f16 elems)

    #pragma unroll 2
    for (int kt = 0; kt < KSTEPS; ++kt) {
        const int cur   = kt & 1;
        const bool more = (kt + 1) < KSTEPS;

        if (more) {                                   // overlaps WMMA below
            const int k1 = (kt + 1) * TILE_K;
            if (MODE == MODE_FULL) {
                asyncA(cur ^ 1, k1);
                asyncB(cur ^ 1, k1);
            } else {
                loadA(k1);
                if (MODE == MODE_PREBIN) asyncB(cur ^ 1, k1); else loadB(k1);
            }
        }
        if (kt + 2 < KSTEPS) {
            const size_t k2 = (size_t)(kt + 2) * TILE_K;
            if (MODE == MODE_FULL) {
                __builtin_prefetch(hTile + (size_t)(tid >> 2) * K_DIM + k2, 0, 0);
                __builtin_prefetch(bTile + (size_t)(tid >> 2) * K_DIM + k2, 0, 0);
            } else {
                __builtin_prefetch(xTile + (size_t)(tid >> 3) * K_DIM + k2, 0, 0);
                if (MODE == MODE_PREBIN)
                    __builtin_prefetch(bTile + (size_t)(tid >> 2) * K_DIM + k2, 0, 0);
                else
                    __builtin_prefetch(wTile + (size_t)(tid >> 3) * K_DIM + k2, 0, 0);
            }
        }

        // ---- fragments from LDS (two contiguous 16B reads per fragment) ----
        v16h aHi[4], aLo[4], bF[4];
        #pragma unroll
        for (int tm = 0; tm < 4; ++tm) {
            const int r = (waveM * 64 + tm * 16 + rsel) * LDSK + kb;
            v8h h0 = *(const v8h*)&sAhi[cur][r];
            v8h h1 = *(const v8h*)&sAhi[cur][r + 16];
            aHi[tm] = __builtin_shufflevector(h0, h1, 0,1,2,3,4,5,6,7,8,9,10,11,12,13,14,15);
            v8h l0 = *(const v8h*)&sAlo[cur][r];
            v8h l1 = *(const v8h*)&sAlo[cur][r + 16];
            aLo[tm] = __builtin_shufflevector(l0, l1, 0,1,2,3,4,5,6,7,8,9,10,11,12,13,14,15);
        }
        #pragma unroll
        for (int tn = 0; tn < 4; ++tn) {
            const int r = (waveN * 64 + tn * 16 + rsel) * LDSK + kb;
            v8h b0 = *(const v8h*)&sB[cur][r];
            v8h b1 = *(const v8h*)&sB[cur][r + 16];
            bF[tn] = __builtin_shufflevector(b0, b1, 0,1,2,3,4,5,6,7,8,9,10,11,12,13,14,15);
        }

        // ---- 32 WMMAs per wave per K-step (hi + lo split) ----
        #pragma unroll
        for (int tm = 0; tm < 4; ++tm) {
            #pragma unroll
            for (int tn = 0; tn < 4; ++tn) {
                acc[tm][tn] = __builtin_amdgcn_wmma_f32_16x16x32_f16(
                    false, aHi[tm], false, bF[tn], (short)0, acc[tm][tn], false, false);
                acc[tm][tn] = __builtin_amdgcn_wmma_f32_16x16x32_f16(
                    false, aLo[tm], false, bF[tn], (short)0, acc[tm][tn], false, false);
            }
        }

        if (more) {
            if (MODE != MODE_FULL) {
                stageA(cur ^ 1);
                if (MODE == MODE_FUSED) stageB(cur ^ 1);
            }
            if (MODE != MODE_FUSED)
                asm volatile("s_wait_asynccnt 0x0" ::: "memory");
        }
        __syncthreads();
    }

    // ---- epilogue: C/D layout -> lane L holds (n = L%16, m = 8*(L/16)+v) ----
    const int colLocal = lane & 15;
    const int rowOff   = (lane >> 4) << 3;
    #pragma unroll
    for (int tn = 0; tn < 4; ++tn) {
        const int col = nBase + waveN * 64 + tn * 16 + colLocal;
        const float bv = bias[col];
        #pragma unroll
        for (int tm = 0; tm < 4; ++tm) {
            const int row0 = mBase + waveM * 64 + tm * 16 + rowOff;
            #pragma unroll
            for (int v = 0; v < 8; ++v)
                out[(size_t)(row0 + v) * N_DIM + col] = acc[tm][tn][v] + bv;
        }
    }
}

extern "C" void kernel_launch(void* const* d_in, const int* in_sizes, int n_in,
                              void* d_out, int out_size, void* d_ws, size_t ws_size,
                              hipStream_t stream) {
    (void)n_in; (void)out_size;
    const float* x    = (const float*)d_in[0];   // [8,2048,4096] fp32
    const float* wgt  = (const float*)d_in[1];   // [4096,4096] fp32
    const float* bias = (const float*)d_in[2];   // [4096] fp32
    float* out = (float*)d_out;

    const int M  = in_sizes[0] / K_DIM;          // 16384
    const int nX = in_sizes[0];                  // 67,108,864
    const int nW = N_DIM * K_DIM;                // 16,777,216
    dim3 grid(M / TILE_M, N_DIM / TILE_N);       // (128, 16)

    const size_t needW = (size_t)nW * sizeof(_Float16);      // 32 MB
    const size_t needX = (size_t)nX * sizeof(_Float16) * 2;  // 256 MB

    if (ws_size >= needW + needX) {
        _Float16* wb  = (_Float16*)d_ws;
        _Float16* xhi = (_Float16*)((char*)d_ws + needW);
        _Float16* xlo = xhi + (size_t)nX;
        binarize_w<<<dim3(nW / (256 * 8)), dim3(256), 0, stream>>>(wgt, wb);
        split_x   <<<dim3(nX / (256 * 8)), dim3(256), 0, stream>>>(x, xhi, xlo);
        binlin_wmma<MODE_FULL><<<grid, dim3(256), 0, stream>>>(
            x, wgt, wb, xhi, xlo, bias, out);
    } else if (ws_size >= needW) {
        _Float16* wb = (_Float16*)d_ws;
        binarize_w<<<dim3(nW / (256 * 8)), dim3(256), 0, stream>>>(wgt, wb);
        binlin_wmma<MODE_PREBIN><<<grid, dim3(256), 0, stream>>>(
            x, wgt, wb, nullptr, nullptr, bias, out);
    } else {
        binlin_wmma<MODE_FUSED><<<grid, dim3(256), 0, stream>>>(
            x, wgt, nullptr, nullptr, nullptr, bias, out);
    }
}
